// WaveNetLayer_25666724561091
// MI455X (gfx1250) — compile-verified, matrix operand order checked by reading, not run on hardware
//
#include <hip/hip_runtime.h>
#include <hip/hip_bf16.h>

typedef __attribute__((ext_vector_type(16))) _Float16 v16h;
typedef __attribute__((ext_vector_type(8)))  _Float16 v8h;
typedef __attribute__((ext_vector_type(8)))  float    v8f;

#define NB   8
#define NC   256
#define NT   8192
#define TOUT 8190          // NT - (K-1)*DIL
#define TN   64            // time tile
#define TROWS (TN + 2)     // rows of input tile (extra 2 for dilation tap)
#define LDH  264           // LDS row pitch in halves (256 + 8 pad -> +4 banks/row)
#define NTILES 128         // ceil(TOUT / TN)
#define SSTRIDE (TROWS * LDH)          // one input stream in LDS (halves)
#define YOFF    (3 * SSTRIDE)          // y tile offset (halves)
#define SMEMH   (YOFF + TN * LDH)      // total dynamic LDS in halves

// ws layout (f16): [0..5]=f-branch segs (x0,x1,c0,c1,g0,g1), [6..11]=g-branch,
// [12]=w_res, [13]=w_skip; each 256x256 row-major [out][in].
#define MATH 65536

__global__ __launch_bounds__(256)
void pack_weights(const float* wfx, const float* wfc, const float* wfg,
                  const float* wgx, const float* wgc, const float* wgg,
                  const float* wres, const float* wskip, _Float16* wh) {
    int idx = blockIdx.x * 256 + threadIdx.x;
    if (idx >= 14 * MATH) return;
    int mat = idx >> 16;
    int r   = idx & (MATH - 1);     // o*256 + i
    float v;
    if (mat < 12) {
        int branch = mat / 6;
        int seg = mat % 6;
        int src = seg >> 1, tap = seg & 1;
        const float* w = (branch == 0)
            ? (src == 0 ? wfx : (src == 1 ? wfc : wfg))
            : (src == 0 ? wgx : (src == 1 ? wgc : wgg));
        v = w[(size_t)r * 2 + tap];            // (o,i,k) -> tap-split
    } else if (mat == 12) {
        v = wres[r];
    } else {
        v = wskip[r];
    }
    wh[idx] = (_Float16)v;
}

__device__ __forceinline__ v16h cat16(v8h lo, v8h hi) {
    return __builtin_shufflevector(lo, hi, 0,1,2,3,4,5,6,7,8,9,10,11,12,13,14,15);
}

// global (weights) fragment
__device__ __forceinline__ v16h ldfragG(const _Float16* p) {
    return cat16(*(const v8h*)p, *(const v8h*)(p + 16));
}

extern __shared__ _Float16 smemH[];   // single LDS root so addrspace(3) is inferred

// LDS fragment: index rooted at the shared symbol -> ds_load_b128
__device__ __forceinline__ v16h ldfragS(int idx) {
    v8h lo = *(const v8h*)(smemH + idx);
    v8h hi = *(const v8h*)(smemH + idx + 16);
    return cat16(lo, hi);
}

__global__ __launch_bounds__(256)
void wavenet_layer(const float* __restrict__ x, const float* __restrict__ cin,
                   const float* __restrict__ gin, const float* __restrict__ skips,
                   const float* __restrict__ bfx, const float* __restrict__ bfc,
                   const float* __restrict__ bfg, const float* __restrict__ bgx,
                   const float* __restrict__ bgc, const float* __restrict__ bgg,
                   const float* __restrict__ bres, const float* __restrict__ bskip,
                   const _Float16* __restrict__ wh, float* __restrict__ out) {
    __shared__ float bL[1024];        // bf[256], bg[256], br[256], bs[256]

    const int tid  = threadIdx.x;
    const int wave = tid >> 5;
    const int lane = tid & 31;
    const int laneLow = lane & 15;
    const int hiSel   = (lane >> 4) & 1;
    const int tile = blockIdx.x;
    const int b    = blockIdx.y;
    const int t0   = tile * TN;
    const size_t outP = (size_t)NB * NC * TOUT;

    // ---- phase 0: stage inputs to LDS (f16, time-major), combined biases ----
    {
        const float* srcs[3] = {x, cin, gin};   // global ptr select only
        for (int s = 0; s < 3; ++s) {
            const float* ps = srcs[s] + (size_t)b * NC * NT;
            const int sOff = s * SSTRIDE;       // integer LDS offset select
            for (int i = tid; i < NC * TROWS; i += 256) {
                int c = i / TROWS;
                int j = i - c * TROWS;
                int t = t0 + j; if (t > NT - 1) t = NT - 1;   // clamp (tail only)
                smemH[sOff + j * LDH + c] = (_Float16)ps[(size_t)c * NT + t];
            }
        }
        // 256 threads, one channel each
        bL[tid]       = bfx[tid] + bfc[tid] + bfg[tid];
        bL[256 + tid] = bgx[tid] + bgc[tid] + bgg[tid];
        bL[512 + tid] = bres[tid];
        bL[768 + tid] = bskip[tid];
    }
    __syncthreads();

    const _Float16* fW = wh;
    const _Float16* gW = wh + 6 * MATH;
    const int fragCol = laneLow * LDH + hiSel * 8;   // per-lane B column base

    // ---- stage 1: pre-activation GEMMs + gating -> Yh (f16) ----
    // B-fragments loaded once per (seg,kt), reused across both M-tiles and
    // both branches: 8 ds_load_b128 feed 16 WMMAs.
    {
        v8f accF[2][4] = {};
        v8f accG[2][4] = {};
        for (int seg = 0; seg < 6; ++seg) {
            const int sBase = (seg >> 1) * SSTRIDE + (seg & 1) * (2 * LDH); // +2 rows = tap
            for (int kt = 0; kt < 8; ++kt) {
                const int kc = kt * 32;
                v16h Bm[4];
#pragma unroll
                for (int nt = 0; nt < 4; ++nt)
                    Bm[nt] = ldfragS(sBase + nt * 16 * LDH + fragCol + kc);
#pragma unroll
                for (int mi = 0; mi < 2; ++mi) {
                    const int mrow = (2 * wave + mi) * 16 + laneLow;
                    v16h Af = ldfragG(fW + ((size_t)seg * 256 + mrow) * 256 + kc + hiSel * 8);
                    v16h Ag = ldfragG(gW + ((size_t)seg * 256 + mrow) * 256 + kc + hiSel * 8);
#pragma unroll
                    for (int nt = 0; nt < 4; ++nt) {
                        accF[mi][nt] = __builtin_amdgcn_wmma_f32_16x16x32_f16(
                            false, Af, false, Bm[nt], (short)0, accF[mi][nt], false, false);
                        accG[mi][nt] = __builtin_amdgcn_wmma_f32_16x16x32_f16(
                            false, Ag, false, Bm[nt], (short)0, accG[mi][nt], false, false);
                    }
                }
            }
        }
        // gating: y = tanh(f+bf) * sigmoid(g+bg); stash to Yh (f16, time-major)
#pragma unroll
        for (int mi = 0; mi < 2; ++mi) {
            const int mt = 2 * wave + mi;
#pragma unroll
            for (int nt = 0; nt < 4; ++nt) {
                const int tl = nt * 16 + laneLow;
#pragma unroll
                for (int r = 0; r < 8; ++r) {
                    const int m = mt * 16 + r + hiSel * 8;
                    float f = accF[mi][nt][r] + bL[m];
                    float g = accG[mi][nt][r] + bL[256 + m];
                    float ef = __expf(2.0f * f);
                    float th = 1.0f - 2.0f * __builtin_amdgcn_rcpf(ef + 1.0f);
                    float sg = __builtin_amdgcn_rcpf(1.0f + __expf(-g));
                    smemH[YOFF + tl * LDH + m] = (_Float16)(th * sg);
                }
            }
        }
    }
    __syncthreads();

    // ---- stage 2: 1x1 res/skip GEMMs + fused residual adds ----
    {
        const _Float16* rW = wh + 12 * MATH;
        const _Float16* sW = wh + 13 * MATH;
        v8f accR[2][4] = {};
        v8f accS[2][4] = {};
        for (int kt = 0; kt < 8; ++kt) {
            const int kc = kt * 32;
            v16h Bm[4];
#pragma unroll
            for (int nt = 0; nt < 4; ++nt)
                Bm[nt] = ldfragS(YOFF + nt * 16 * LDH + fragCol + kc);
#pragma unroll
            for (int mi = 0; mi < 2; ++mi) {
                const int mrow = (2 * wave + mi) * 16 + laneLow;
                v16h Ar = ldfragG(rW + (size_t)mrow * 256 + kc + hiSel * 8);
                v16h As = ldfragG(sW + (size_t)mrow * 256 + kc + hiSel * 8);
#pragma unroll
                for (int nt = 0; nt < 4; ++nt) {
                    accR[mi][nt] = __builtin_amdgcn_wmma_f32_16x16x32_f16(
                        false, Ar, false, Bm[nt], (short)0, accR[mi][nt], false, false);
                    accS[mi][nt] = __builtin_amdgcn_wmma_f32_16x16x32_f16(
                        false, As, false, Bm[nt], (short)0, accS[mi][nt], false, false);
                }
            }
        }
#pragma unroll
        for (int mi = 0; mi < 2; ++mi) {
            const int mt = 2 * wave + mi;
#pragma unroll
            for (int nt = 0; nt < 4; ++nt) {
                const int tl = nt * 16 + laneLow;
                const int tg = t0 + tl;
                if (tg < TOUT) {
#pragma unroll
                    for (int r = 0; r < 8; ++r) {
                        const int m = mt * 16 + r + hiSel * 8;
                        const size_t iidx = ((size_t)b * NC + m) * NT + tg + 2;
                        const size_t oidx = ((size_t)b * NC + m) * TOUT + tg;
                        out[oidx]            = accR[mi][nt][r] + bL[512 + m] + x[iidx];
                        out[3 * outP + oidx] = accS[mi][nt][r] + bL[768 + m] + skips[iidx];
                    }
                }
            }
        }
    }

    // ---- cin/gin sliced passthrough (pure f32 copies) ----
    for (int i = tid; i < NC * TN; i += 256) {
        const int c = i >> 6, tl = i & 63;
        const int tg = t0 + tl;
        if (tg < TOUT) {
            const size_t src = ((size_t)b * NC + c) * NT + tg + 2;
            const size_t dst = ((size_t)b * NC + c) * TOUT + tg;
            out[outP + dst]     = cin[src];
            out[2 * outP + dst] = gin[src];
        }
    }
}

extern "C" void kernel_launch(void* const* d_in, const int* in_sizes, int n_in,
                              void* d_out, int out_size, void* d_ws, size_t ws_size,
                              hipStream_t stream) {
    const float* x     = (const float*)d_in[0];
    const float* cin   = (const float*)d_in[1];
    const float* gin   = (const float*)d_in[2];
    const float* skips = (const float*)d_in[3];
    const float* w_fx  = (const float*)d_in[4];
    const float* b_fx  = (const float*)d_in[5];
    const float* w_fc  = (const float*)d_in[6];
    const float* b_fc  = (const float*)d_in[7];
    const float* w_fg  = (const float*)d_in[8];
    const float* b_fg  = (const float*)d_in[9];
    const float* w_gx  = (const float*)d_in[10];
    const float* b_gx  = (const float*)d_in[11];
    const float* w_gc  = (const float*)d_in[12];
    const float* b_gc  = (const float*)d_in[13];
    const float* w_gg  = (const float*)d_in[14];
    const float* b_gg  = (const float*)d_in[15];
    const float* w_res = (const float*)d_in[16];
    const float* b_res = (const float*)d_in[17];
    const float* w_skip= (const float*)d_in[18];
    const float* b_skip= (const float*)d_in[19];
    float* out = (float*)d_out;
    _Float16* wh = (_Float16*)d_ws;   // 14*65536 halves = 1.75 MB

    // 1) pack all weights to f16 in workspace
    {
        int total = 14 * MATH;
        pack_weights<<<(total + 255) / 256, 256, 0, stream>>>(
            w_fx, w_fc, w_fg, w_gx, w_gc, w_gg, w_res, w_skip, wh);
    }
    // 2) fused WaveNet layer
    {
        dim3 grid(NTILES, NB);
        size_t shmem = (size_t)SMEMH * sizeof(_Float16);   // 138336 B dynamic
        wavenet_layer<<<grid, 256, shmem, stream>>>(
            x, cin, gin, skips,
            b_fx, b_fc, b_fg, b_gx, b_gc, b_gg, b_res, b_skip,
            wh, out);
    }
}
